// RNNSTSPCell_20787641712912
// MI455X (gfx1250) — compile-verified, matrix operand order checked by reading, not run on hardware
//
#include <hip/hip_runtime.h>
#include <hip/hip_bf16.h>

// ---------------------------------------------------------------------------
// Problem constants (from reference)
// ---------------------------------------------------------------------------
#define BATCH   4096
#define N_IN    512
#define N_HID   2048
#define K_TOT   2560          // N_IN + N_HID (fused GEMM K)
#define DT_SEC  0.01f
#define ALPHA_N 0.2f

typedef __bf16 bf16_t;
typedef __attribute__((ext_vector_type(16))) __bf16 v16bf;
typedef __attribute__((ext_vector_type(8)))  __bf16 v8bf;
typedef __attribute__((ext_vector_type(8)))  float  v8f;
typedef __attribute__((ext_vector_type(4)))  unsigned int u32x4;
typedef __attribute__((ext_vector_type(8)))  int i32x8;
typedef __attribute__((ext_vector_type(4)))  int i32x4;

// ---------------------------------------------------------------------------
// Feature detection: Tensor Data Mover path
// ---------------------------------------------------------------------------
#if defined(__has_builtin)
#if __has_builtin(__builtin_amdgcn_tensor_load_to_lds) && \
    __has_builtin(__builtin_amdgcn_s_wait_tensorcnt)
#define HAVE_TDM 1
#endif
#endif
#ifndef HAVE_TDM
#define HAVE_TDM 0
#endif

// ---------------------------------------------------------------------------
// GEMM tiling
// ---------------------------------------------------------------------------
#define TILE_M 128
#define TILE_N 128
#define TILE_K 32
#define LDSS   40                       // 32 + 8 pad elems; 80B row stride (16B aligned)
#define NSTAGES (K_TOT / TILE_K)        // 80
#define LDS_W_OFF (TILE_M * LDSS)       // element offset of W region inside a buffer

// Build a 16x32 bf16 WMMA fragment (A-layout / B^T-layout) from LDS.
// Lane l: row = base + (l&15); k-half selected by (l>>4).
// lo = 8 bf16 at k = half*8, hi = 8 bf16 at k = 16 + half*8.
__device__ __forceinline__ v16bf frag_ld(const bf16_t* p) {
    v8bf lo = *(const v8bf*)(p);
    v8bf hi = *(const v8bf*)(p + 16);
    return __builtin_shufflevector(lo, hi, 0,1,2,3,4,5,6,7,8,9,10,11,12,13,14,15);
}

#if HAVE_TDM
// ---------------------------------------------------------------------------
// TDM 2-D tile load: rows of `tile_d0` bf16 elements, `tile_d1` rows, row
// stride `d0_stride` elements in global memory. LDS destination gets 16B of
// padding after every 64B row (pad_interval=16 DW, pad_amount=4 DW), i.e. an
// effective LDS row stride of LDSS=40 elements.
// Toolchain arity differs: ROCm 7.2 (clang-22) = 5 args; amdgpu-toolchain
// (clang-23+) = 6 args with an extra int32x8 before cpol.
// ---------------------------------------------------------------------------
__device__ __forceinline__ void tdm_load_2d(
    unsigned lds_addr, unsigned long long gaddr,
    unsigned tensor_d0, unsigned tensor_d1, unsigned long long d0_stride,
    unsigned tile_d0, unsigned tile_d1)
{
    u32x4 g0;
    g0[0] = 1u;                                            // count=1, user descriptor
    g0[1] = lds_addr;                                      // D#[63:32]  lds_addr
    g0[2] = (unsigned)(gaddr & 0xffffffffu);               // D#[95:64]  global_addr lo
    g0[3] = (unsigned)((gaddr >> 32) & 0x01ffffffu)        // D#[120:96] global_addr hi
          | (2u << 30);                                    // D#[127:126] type = 2
    i32x8 g1;
    g1[0] = (1 << 16)                                      // data_size = 2 bytes
          | (1 << 20)                                      // pad_enable
          | (3 << 22)                                      // pad_interval: 16 DWORDs
          | (3 << 25);                                     // pad_amount : 4 DWORDs
    g1[1] = (int)((tensor_d0 & 0xffffu) << 16);            // tensor_dim0[15:0]
    g1[2] = (int)((tensor_d0 >> 16) |
                  ((tensor_d1 & 0xffffu) << 16));          // dim0 hi | dim1 lo
    g1[3] = (int)((tensor_d1 >> 16) | (tile_d0 << 16));    // dim1 hi | tile_dim0
    g1[4] = (int)(tile_d1 & 0xffffu);                      // tile_dim1 (tile_dim2 = 0)
    g1[5] = (int)(d0_stride & 0xffffffffu);                // tensor_dim0_stride lo
    g1[6] = (int)((d0_stride >> 32) & 0xffffu);            // stride hi (dim1_stride = 0)
    g1[7] = 0;
    i32x4 z4 = {0, 0, 0, 0};
#if __clang_major__ >= 23
    i32x8 z8 = {0, 0, 0, 0, 0, 0, 0, 0};
    __builtin_amdgcn_tensor_load_to_lds(g0, g1, z4, z4, z8, 0);
#else
    __builtin_amdgcn_tensor_load_to_lds(g0, g1, z4, z4, 0);
#endif
}
#endif

// ---------------------------------------------------------------------------
// Kernel 1: weight prep.  Wt[n][k] (bf16, N-major, K contiguous):
//   k <  512 : relu(W_ih[k][n])
//   k >= 512 : mask[kk][n] * ei[kk] * relu(W_hh[kk][n]),  kk = k-512
// ---------------------------------------------------------------------------
__global__ __launch_bounds__(256) void wprep_kernel(
    const float* __restrict__ w_ih, const float* __restrict__ w_hh,
    const float* __restrict__ ei,   const float* __restrict__ mask,
    bf16_t* __restrict__ Wt)
{
    int idx = blockIdx.x * 256 + threadIdx.x;     // over K_TOT * N_HID
    int n = idx & (N_HID - 1);
    int k = idx >> 11;
    float v;
    if (k < N_IN) {
        v = w_ih[idx];                            // idx == k*N_HID + n
        v = v > 0.f ? v : 0.f;
    } else {
        int kk = k - N_IN;
        size_t o = (size_t)kk * N_HID + n;
        float w = w_hh[o];
        w = w > 0.f ? w : 0.f;
        v = mask[o] * ei[kk] * w;
    }
    Wt[(size_t)n * K_TOT + k] = (bf16_t)v;
}

// ---------------------------------------------------------------------------
// Kernel 2: inp -> bf16 into fused A matrix, columns [0, 512)
// ---------------------------------------------------------------------------
__global__ __launch_bounds__(256) void inpconv_kernel(
    const float* __restrict__ inp, bf16_t* __restrict__ Abf)
{
    int idx = blockIdx.x * 256 + threadIdx.x;     // over BATCH * N_IN
    int i = idx & (N_IN - 1);
    int b = idx >> 9;
    Abf[(size_t)b * K_TOT + i] = (bf16_t)inp[idx];
}

// ---------------------------------------------------------------------------
// Kernel 3: STSP elementwise update. Writes syn_x/syn_u outputs (f32) and
// h_post (bf16) into fused A matrix columns [512, 2560).
// ---------------------------------------------------------------------------
__global__ __launch_bounds__(256) void stsp_kernel(
    const float* __restrict__ h_in,  const float* __restrict__ syn_x,
    const float* __restrict__ syn_u, const float* __restrict__ a_stf,
    const float* __restrict__ a_std, const float* __restrict__ Uv,
    const float* __restrict__ dynsyn,
    float* __restrict__ out_synx, float* __restrict__ out_synu,
    bf16_t* __restrict__ Abf)
{
    int idx = blockIdx.x * 256 + threadIdx.x;     // over BATCH * N_HID
    int j = idx & (N_HID - 1);
    int b = idx >> 11;
    float h  = h_in[idx];
    float sx = syn_x[idx];
    float su = syn_u[idx];
    float ds = dynsyn[j];
    float u  = Uv[j];
    float nsx = sx + (a_std[j] * (1.f - sx) - DT_SEC * su * sx * h) * ds;
    float nsu = su + (a_stf[j] * (u - su) + DT_SEC * u * (1.f - su) * h) * ds;
    nsx = fminf(fmaxf(nsx, 0.f), 1.f);
    nsu = fminf(fmaxf(nsu, 0.f), 1.f);
    out_synx[idx] = nsx;
    out_synu[idx] = nsu;
    Abf[(size_t)b * K_TOT + N_IN + j] = (bf16_t)(nsu * nsx * h);
}

// ---------------------------------------------------------------------------
// Kernel 4: fused bf16 WMMA GEMM + epilogue, TDM double-buffered pipeline.
//   pre = A(4096x2560) @ Wt^T + bias ; h_out = 0.8*h_in + 0.2*relu(pre)
// Block = 256 threads (8 waves), C tile 128x128, wave owns 32x64 (2x4 WMMA).
// Wave 0 drives the Tensor Data Mover: stage s+1 DMAs into buffer (s+1)&1
// while all waves compute stage s from buffer s&1 (TENSORcnt + barriers).
// ---------------------------------------------------------------------------
__global__ __launch_bounds__(256) void gemm_kernel(
    const bf16_t* __restrict__ A,    // [BATCH, K_TOT]
    const bf16_t* __restrict__ W,    // [N_HID, K_TOT]  (= W^T, N-major)
    const float*  __restrict__ bias, // [N_HID]
    const float*  __restrict__ h_in, // [BATCH, N_HID]
    float*        __restrict__ out)  // [BATCH, N_HID]  (h_out region)
{
    __shared__ __align__(16) bf16_t lds[2][(TILE_M + TILE_N) * LDSS];

    const int bn0  = blockIdx.x * TILE_N;
    const int bm0  = blockIdx.y * TILE_M;
    const int tid  = threadIdx.x;
    const int lane = tid & 31;
    const int wave = tid >> 5;
    const int wm   = wave & 3;      // 0..3 : 32 rows each
    const int wn   = wave >> 2;     // 0..1 : 64 cols each
    const int half = lane >> 4;
    const int l16  = lane & 15;

    v8f acc[2][4];
    #pragma unroll
    for (int mt = 0; mt < 2; ++mt)
        #pragma unroll
        for (int nt = 0; nt < 4; ++nt)
            #pragma unroll
            for (int r = 0; r < 8; ++r)
                acc[mt][nt][r] = 0.0f;

#if HAVE_TDM
    // Prologue: DMA stage 0 into buffer 0.
    if (wave == 0) {
        unsigned ldsA = (unsigned)(size_t)&lds[0][0];          // flat->LDS aperture: low 32b
        tdm_load_2d(ldsA,
                    (unsigned long long)(size_t)(A + (size_t)bm0 * K_TOT),
                    K_TOT, BATCH, K_TOT, TILE_K, TILE_M);
        tdm_load_2d(ldsA + LDS_W_OFF * 2,
                    (unsigned long long)(size_t)(W + (size_t)bn0 * K_TOT),
                    K_TOT, N_HID, K_TOT, TILE_K, TILE_N);
    }
#endif

    for (int s = 0; s < NSTAGES; ++s) {
        const int cur = s & 1;
#if HAVE_TDM
        if (wave == 0) {
            if (s + 1 < NSTAGES) {
                const int k0 = (s + 1) * TILE_K;
                unsigned ldsA = (unsigned)(size_t)&lds[cur ^ 1][0];
                tdm_load_2d(ldsA,
                            (unsigned long long)(size_t)(A + (size_t)bm0 * K_TOT + k0),
                            K_TOT, BATCH, K_TOT, TILE_K, TILE_M);
                tdm_load_2d(ldsA + LDS_W_OFF * 2,
                            (unsigned long long)(size_t)(W + (size_t)bn0 * K_TOT + k0),
                            K_TOT, N_HID, K_TOT, TILE_K, TILE_N);
                __builtin_amdgcn_s_wait_tensorcnt(2);  // stage s's pair retired
            } else {
                __builtin_amdgcn_s_wait_tensorcnt(0);  // last stage: drain
            }
        }
        __syncthreads();                                // release stage s to all waves
        const bf16_t* buf = &lds[cur][0];
#else
        // Fallback: synchronous staging through VGPRs, single buffer.
        __syncthreads();
        {
            const int k0  = s * TILE_K;
            const int row = tid >> 2;          // 0..63 (+64 second pass)
            const int col = (tid & 3) * 8;     // 8 bf16 = 16B per thread
            #pragma unroll
            for (int p = 0; p < 2; ++p) {
                int r = row + p * 64;
                *(uint4*)(&lds[0][r * LDSS + col]) =
                    *(const uint4*)(A + (size_t)(bm0 + r) * K_TOT + k0 + col);
                *(uint4*)(&lds[0][LDS_W_OFF + r * LDSS + col]) =
                    *(const uint4*)(W + (size_t)(bn0 + r) * K_TOT + k0 + col);
            }
        }
        __syncthreads();
        const bf16_t* buf = &lds[0][0];
#endif

        // One 16x16x32 k-step per stage.
        v16bf af[2], bfrag[4];
        #pragma unroll
        for (int mt = 0; mt < 2; ++mt)
            af[mt] = frag_ld(buf + (wm * 32 + mt * 16 + l16) * LDSS + half * 8);
        #pragma unroll
        for (int nt = 0; nt < 4; ++nt)
            bfrag[nt] = frag_ld(buf + LDS_W_OFF
                                + (wn * 64 + nt * 16 + l16) * LDSS + half * 8);
        #pragma unroll
        for (int mt = 0; mt < 2; ++mt)
            #pragma unroll
            for (int nt = 0; nt < 4; ++nt)
                acc[mt][nt] = __builtin_amdgcn_wmma_f32_16x16x32_bf16(
                    false, af[mt], false, bfrag[nt],
                    (short)0, acc[mt][nt], false, false);

#if HAVE_TDM
        __syncthreads();   // all reads of buf done before TDM refills it next iter
#endif
    }

    // Epilogue: pre = acc + bias ; h_out = 0.8*h_in + 0.2*relu(pre)
    #pragma unroll
    for (int nt = 0; nt < 4; ++nt) {
        const int n = bn0 + wn * 64 + nt * 16 + l16;
        const float bz = bias[n];
        #pragma unroll
        for (int mt = 0; mt < 2; ++mt) {
            #pragma unroll
            for (int r = 0; r < 8; ++r) {
                const int m = bm0 + wm * 32 + mt * 16 + half * 8 + r;
                const size_t o = (size_t)m * N_HID + n;
                float pre = acc[mt][nt][r] + bz;
                float rel = pre > 0.f ? pre : 0.f;
                out[o] = h_in[o] * (1.0f - ALPHA_N) + ALPHA_N * rel;
            }
        }
    }
}

// ---------------------------------------------------------------------------
// Launch
// ---------------------------------------------------------------------------
extern "C" void kernel_launch(void* const* d_in, const int* in_sizes, int n_in,
                              void* d_out, int out_size, void* d_ws, size_t ws_size,
                              hipStream_t stream) {
    const float* inp    = (const float*)d_in[0];
    const float* h_in   = (const float*)d_in[1];
    const float* syn_x  = (const float*)d_in[2];
    const float* syn_u  = (const float*)d_in[3];
    const float* w_ih   = (const float*)d_in[4];
    const float* w_hh   = (const float*)d_in[5];
    const float* bias   = (const float*)d_in[6];
    const float* ei     = (const float*)d_in[7];
    const float* mask   = (const float*)d_in[8];
    const float* a_stf  = (const float*)d_in[9];
    const float* a_std  = (const float*)d_in[10];
    const float* Uv     = (const float*)d_in[11];
    const float* dynsyn = (const float*)d_in[12];

    float* out_h    = (float*)d_out;                         // h_out
    float* out_synx = out_h + (size_t)BATCH * N_HID;         // syn_x
    float* out_synu = out_synx + (size_t)BATCH * N_HID;      // syn_u

    bf16_t* Abf = (bf16_t*)d_ws;                                       // [BATCH, K_TOT]
    bf16_t* Wt  = (bf16_t*)((char*)d_ws + (size_t)BATCH * K_TOT * 2);  // [N_HID, K_TOT]

    wprep_kernel<<<(K_TOT * N_HID) / 256, 256, 0, stream>>>(w_ih, w_hh, ei, mask, Wt);
    inpconv_kernel<<<(BATCH * N_IN) / 256, 256, 0, stream>>>(inp, Abf);
    stsp_kernel<<<(BATCH * N_HID) / 256, 256, 0, stream>>>(
        h_in, syn_x, syn_u, a_stf, a_std, Uv, dynsyn, out_synx, out_synu, Abf);

    dim3 grid(N_HID / TILE_N, BATCH / TILE_M);
    gemm_kernel<<<grid, 256, 0, stream>>>(Abf, Wt, bias, h_in, out_h);
}
